// SimpleMemory_38611755991503
// MI455X (gfx1250) — matrix operand order repaired and checked by reading, not compiled, not run on hardware
//
#include <hip/hip_runtime.h>
#include <hip/hip_bf16.h>
#include <math.h>

// ---------------- problem constants (from reference setup_inputs) ----------
static constexpr int B   = 8;
static constexpr int S   = 2048;
static constexpr int H   = 1024;
static constexpr int CAP = 65536;
static constexpr int TK  = 16;

// ---------------- WMMA vector types ---------------------------------------
typedef __attribute__((ext_vector_type(8)))  float          v8f;
typedef __attribute__((ext_vector_type(16))) __bf16         v16bf;
typedef __attribute__((ext_vector_type(16))) unsigned short v16u;

static __device__ __forceinline__ unsigned short f2bf(float x) {
    unsigned u = __builtin_bit_cast(unsigned, x);
    unsigned r = u + 0x7FFFu + ((u >> 16) & 1u);   // round-to-nearest-even
    return (unsigned short)(r >> 16);
}

// ---------------- K1: q_pooled = mean_s(query) -----------------------------
__global__ __launch_bounds__(256) void k_pool(const float* __restrict__ q,
                                              float* __restrict__ qpool) {
    const int b = blockIdx.x;
    const int h = blockIdx.y * 256 + threadIdx.x;
    const float* p = q + (size_t)b * S * H + h;
    float acc = 0.f;
    for (int s = 0; s < S; ++s) acc += p[(size_t)s * H];
    qpool[b * H + h] = acc * (1.0f / (float)S);
}

// ---------------- K2: qp = q_pooled @ W^T  (one wave per output) -----------
__global__ __launch_bounds__(256) void k_qp(const float* __restrict__ qpool,
                                            const float* __restrict__ w,
                                            float* __restrict__ qp) {
    const int b    = blockIdx.x;
    const int wv   = threadIdx.x >> 5;
    const int lane = threadIdx.x & 31;
    const int i    = blockIdx.y * 8 + wv;
    float acc = 0.f;
    for (int h = lane; h < H; h += 32)
        acc += qpool[b * H + h] * w[(size_t)i * H + h];
    for (int m = 16; m; m >>= 1) acc += __shfl_xor(acc, m, 32);
    if (lane == 0) qp[b * H + i] = acc;
}

// ---------------- K3: qhat = l2norm(qp) per row ----------------------------
__global__ __launch_bounds__(256) void k_qnorm(const float* __restrict__ qp,
                                               float* __restrict__ qhat) {
    const int b = blockIdx.x;
    __shared__ float red[256];
    const int t = threadIdx.x;
    float ss = 0.f;
    for (int h = t; h < H; h += 256) { float v = qp[b * H + h]; ss += v * v; }
    red[t] = ss; __syncthreads();
    for (int off = 128; off; off >>= 1) {
        if (t < off) red[t] += red[t + off];
        __syncthreads();
    }
    const float sc = 1.0f / fmaxf(sqrtf(red[0]), 1e-12f);
    for (int h = t; h < H; h += 256) qhat[b * H + h] = qp[b * H + h] * sc;
}

// ---------------- K4: sims[b,c] = qhat[b] . keys[c]/||keys[c]|| ------------
// One wave per key row; key read exactly once, norm + 8 batch dots fused.
__global__ __launch_bounds__(256) void k_sims(const float* __restrict__ keys,
                                              const float* __restrict__ qhat,
                                              float* __restrict__ sims) {
    __shared__ float qsh[B * H];                       // 32 KB
    const int t = threadIdx.x;
    for (int i = t; i < B * H; i += 256) qsh[i] = qhat[i];
    __syncthreads();
    const int wv   = t >> 5;
    const int lane = t & 31;
    const int c    = blockIdx.x * 8 + wv;
    const float* krow = keys + (size_t)c * H;
    float sq = 0.f, dot[B];
#pragma unroll
    for (int bb = 0; bb < B; ++bb) dot[bb] = 0.f;
    for (int h = lane; h < H; h += 32) {
        float kx = krow[h];
        sq += kx * kx;
#pragma unroll
        for (int bb = 0; bb < B; ++bb) dot[bb] += kx * qsh[bb * H + h];
    }
    for (int m = 16; m; m >>= 1) {
        sq += __shfl_xor(sq, m, 32);
#pragma unroll
        for (int bb = 0; bb < B; ++bb) dot[bb] += __shfl_xor(dot[bb], m, 32);
    }
    if (lane == 0) {
        const float rn = 1.0f / fmaxf(sqrtf(sq), 1e-12f);
#pragma unroll
        for (int bb = 0; bb < B; ++bb) sims[(size_t)bb * CAP + c] = dot[bb] * rn;
    }
}

// ---------------- K5: top-16 per batch (iterative argmax, no mutation) -----
__global__ __launch_bounds__(256) void k_topk(const float* __restrict__ sims,
                                              int* __restrict__ tidx,
                                              float* __restrict__ tval) {
    const int b = blockIdx.x;
    __shared__ float bv[256];
    __shared__ int   bi[256];
    __shared__ int   chosen[TK];
    const int t = threadIdx.x;
    const float* srow = sims + (size_t)b * CAP;
    for (int k = 0; k < TK; ++k) {
        float lv = -3.4e38f; int li = CAP;
        for (int c = t; c < CAP; c += 256) {
            float v = srow[c];
            bool skip = false;
            for (int j = 0; j < k; ++j) skip |= (chosen[j] == c);
            if (!skip && (v > lv || (v == lv && c < li))) { lv = v; li = c; }
        }
        bv[t] = lv; bi[t] = li; __syncthreads();
        for (int off = 128; off; off >>= 1) {
            if (t < off) {
                if (bv[t + off] > bv[t] ||
                    (bv[t + off] == bv[t] && bi[t + off] < bi[t])) {
                    bv[t] = bv[t + off]; bi[t] = bi[t + off];
                }
            }
            __syncthreads();
        }
        if (t == 0) {
            chosen[k] = bi[0];
            tidx[b * TK + k] = bi[0];
            tval[b * TK + k] = bv[0];
        }
        __syncthreads();
    }
}

// ---------------- K6a: flag = (max top score < 0) ? 0 : 1 ------------------
__global__ __launch_bounds__(128) void k_flag(const float* __restrict__ tval,
                                              float* __restrict__ flag) {
    __shared__ float red[128];
    const int t = threadIdx.x;
    red[t] = tval[t]; __syncthreads();
    for (int off = 64; off; off >>= 1) {
        if (t < off) red[t] = fmaxf(red[t], red[t + off]);
        __syncthreads();
    }
    if (t == 0) flag[0] = (red[0] < 0.0f) ? 0.0f : 1.0f;
}

// ---------------- K6b: gather top-k value rows -----------------------------
__global__ __launch_bounds__(256) void k_gather(const float* __restrict__ values,
                                                const int* __restrict__ tidx,
                                                float* __restrict__ valsg) {
    const int bk = blockIdx.x;
    const int h  = blockIdx.y * 256 + threadIdx.x;
    const int src = tidx[bk];
    valsg[(size_t)bk * H + h] = values[(size_t)src * H + h];
}

// ---------------- K6c: wv[b,k,h] = sum_i vals[b,k,i] * w[i,h] --------------
__global__ __launch_bounds__(256) void k_wv(const float* __restrict__ valsg,
                                            const float* __restrict__ w,
                                            float* __restrict__ wvout) {
    __shared__ float vrow[H];                          // 4 KB
    const int bk = blockIdx.x;
    const int t  = threadIdx.x;
    for (int i = t; i < H; i += 256) vrow[i] = valsg[(size_t)bk * H + i];
    __syncthreads();
    const int h = blockIdx.y * 256 + t;
    float acc = 0.f;
    for (int i = 0; i < H; ++i) acc += vrow[i] * w[(size_t)i * H + h];
    wvout[(size_t)bk * H + h] = acc;
}

// ---------------- K7: fused scores->softmax->out, WMMA bf16 ---------------
// One wave per 16-row S-chunk. 4 waves per block.
// Phase 1: C[s,k] = sum_h Q[s,h]*WV[k,h]  (32x v_wmma_f32_16x16x32_bf16)
// Phase 2: softmax over k (cross-lane within 16-lane halves = C layout rows)
// Phase 3: out[s,h] = sum_k attn[s,k]*vals[k,h], K padded 16->32 with zeros
__global__ __launch_bounds__(128) void k_attend(const float* __restrict__ query,
                                                const float* __restrict__ wv,
                                                const float* __restrict__ valsg,
                                                const float* __restrict__ flag,
                                                float* __restrict__ out) {
    const int b    = blockIdx.x;
    const int lane = threadIdx.x;           // 0..31
    const int ty   = threadIdx.y;           // 0..3
    const int s0   = (blockIdx.y * 4 + ty) * 16;

    __shared__ float attn_s[4][16][16];     // per-wave attn tile, 4 KB

    const int row = lane & 15;              // A: M row; B: N col; C: N col
    const int hi  = (lane >> 4) & 1;        // half-wave selector
    const int klo = hi * 8;                 // A-fragment K offset within 16

    const float* qrow  = query + (size_t)(b * S + s0 + row) * H;
    const float* wvrow = wv    + (size_t)(b * TK + row) * H;     // row = k col

    // ---- Phase 1: attention scores -------------------------------------
    v8f c = {};
    for (int h0 = 0; h0 < H; h0 += 32) {
        // A fragment: Q[row, h0+klo+{0..7}] and Q[row, h0+16+klo+{0..7}]
        const float4* pa0 = (const float4*)(qrow + h0 + klo);
        const float4* pa1 = (const float4*)(qrow + h0 + 16 + klo);
        float4 a0 = pa0[0], a1 = pa0[1], a2 = pa1[0], a3 = pa1[1];
        v16u au;
        au[0]=f2bf(a0.x); au[1]=f2bf(a0.y); au[2]=f2bf(a0.z); au[3]=f2bf(a0.w);
        au[4]=f2bf(a1.x); au[5]=f2bf(a1.y); au[6]=f2bf(a1.z); au[7]=f2bf(a1.w);
        au[8]=f2bf(a2.x); au[9]=f2bf(a2.y); au[10]=f2bf(a2.z); au[11]=f2bf(a2.w);
        au[12]=f2bf(a3.x); au[13]=f2bf(a3.y); au[14]=f2bf(a3.z); au[15]=f2bf(a3.w);
        // B fragment: WV[col, h0+klo+{0..7}] and WV[col, h0+16+klo+{0..7}]
        const float4* pb0 = (const float4*)(wvrow + h0 + klo);
        const float4* pb1 = (const float4*)(wvrow + h0 + 16 + klo);
        float4 b0 = pb0[0], b1 = pb0[1], b2 = pb1[0], b3 = pb1[1];
        v16u bu;
        bu[0]=f2bf(b0.x); bu[1]=f2bf(b0.y); bu[2]=f2bf(b0.z); bu[3]=f2bf(b0.w);
        bu[4]=f2bf(b1.x); bu[5]=f2bf(b1.y); bu[6]=f2bf(b1.z); bu[7]=f2bf(b1.w);
        bu[8]=f2bf(b2.x); bu[9]=f2bf(b2.y); bu[10]=f2bf(b2.z); bu[11]=f2bf(b2.w);
        bu[12]=f2bf(b3.x); bu[13]=f2bf(b3.y); bu[14]=f2bf(b3.z); bu[15]=f2bf(b3.w);
        c = __builtin_amdgcn_wmma_f32_16x16x32_bf16(
                false, __builtin_bit_cast(v16bf, au),
                false, __builtin_bit_cast(v16bf, bu),
                (short)0, c, false, false);
    }

    // ---- Phase 2: softmax over k (16 lanes of each half hold one row) ---
    const float scale = flag[0];
#pragma unroll
    for (int v = 0; v < 8; ++v) {
        float m = c[v];
        m = fmaxf(m, __shfl_xor(m, 1, 32));
        m = fmaxf(m, __shfl_xor(m, 2, 32));
        m = fmaxf(m, __shfl_xor(m, 4, 32));
        m = fmaxf(m, __shfl_xor(m, 8, 32));
        float e = __expf(c[v] - m);
        float sm = e;
        sm += __shfl_xor(sm, 1, 32);
        sm += __shfl_xor(sm, 2, 32);
        sm += __shfl_xor(sm, 4, 32);
        sm += __shfl_xor(sm, 8, 32);
        attn_s[ty][v + hi * 8][row] = e / sm;   // [s-row][k-col]
    }
    __syncthreads();

    // ---- Phase 3: out = attn @ vals (K=16 padded to 32 with zeros) ------
    v16u a2u;
#pragma unroll
    for (int j = 0; j < 8; ++j) a2u[j] = f2bf(attn_s[ty][row][klo + j]);
#pragma unroll
    for (int j = 8; j < 16; ++j) a2u[j] = 0;   // K=16..31 zero pad
    const v16bf a2 = __builtin_bit_cast(v16bf, a2u);

    for (int n0 = 0; n0 < H; n0 += 16) {
        v16u b2u;
#pragma unroll
        for (int j = 0; j < 16; ++j) {
            // col=row; j<8: real k = klo+j; j>=8: K=16..31 zeros
            float x = (j < 8) ? valsg[(size_t)(b * TK + klo + j) * H + n0 + row]
                              : 0.0f;
            b2u[j] = f2bf(x);
        }
        v8f c2 = {};
        c2 = __builtin_amdgcn_wmma_f32_16x16x32_bf16(
                 false, a2,
                 false, __builtin_bit_cast(v16bf, b2u),
                 (short)0, c2, false, false);
#pragma unroll
        for (int v = 0; v < 8; ++v)
            out[(size_t)(b * S + s0 + v + hi * 8) * H + n0 + row] = c2[v] * scale;
    }
}

// ---------------- launcher -------------------------------------------------
extern "C" void kernel_launch(void* const* d_in, const int* in_sizes, int n_in,
                              void* d_out, int out_size, void* d_ws, size_t ws_size,
                              hipStream_t stream) {
    (void)in_sizes; (void)n_in; (void)out_size; (void)ws_size;
    const float* query  = (const float*)d_in[0];   // [B,S,H]
    const float* keys   = (const float*)d_in[1];   // [CAP,H]
    const float* values = (const float*)d_in[2];   // [CAP,H]
    const float* wq     = (const float*)d_in[3];   // [H,H]
    // d_in[4] = top_k (==16, matches WMMA tile width; hardcoded)
    float* out = (float*)d_out;

    // workspace layout (floats)
    float* ws     = (float*)d_ws;
    float* qpool  = ws;                 // 8192
    float* qp     = ws + 8192;          // 8192
    float* qhat   = ws + 16384;         // 8192
    float* sims   = ws + 24576;         // 524288
    float* valsg  = ws + 548864;        // 131072
    float* wvbuf  = ws + 679936;        // 131072
    float* tval   = ws + 811008;        // 128
    float* flag   = ws + 811136;        // 1
    int*   tidx   = (int*)(ws + 811168);// 128 ints

    k_pool  <<<dim3(B, H / 256),      256,          0, stream>>>(query, qpool);
    k_qp    <<<dim3(B, H / 8),        256,          0, stream>>>(qpool, wq, qp);
    k_qnorm <<<dim3(B),               256,          0, stream>>>(qp, qhat);
    k_sims  <<<dim3(CAP / 8),         256,          0, stream>>>(keys, qhat, sims);
    k_topk  <<<dim3(B),               256,          0, stream>>>(sims, tidx, tval);
    k_flag  <<<dim3(1),               128,          0, stream>>>(tval, flag);
    k_gather<<<dim3(B * TK, H / 256), 256,          0, stream>>>(values, tidx, valsg);
    k_wv    <<<dim3(B * TK, H / 256), 256,          0, stream>>>(valsg, wq, wvbuf);
    k_attend<<<dim3(B, S / 16 / 4),   dim3(32, 4),  0, stream>>>(query, wvbuf, valsg, flag, out);
}
